// MEMORYVIT_13202729468384
// MI455X (gfx1250) — matrix operand order repaired and checked by reading, not compiled, not run on hardware
//
#include <hip/hip_runtime.h>
#include <hip/hip_bf16.h>

typedef __attribute__((ext_vector_type(16))) _Float16 v16h;
typedef __attribute__((ext_vector_type(8)))  _Float16 v8h;
typedef __attribute__((ext_vector_type(8)))  float    v8f;

#define SCALEF 0.08838834764831845f   // 1/sqrt(128)
#define VPAD 224                      // padded token stride for vT (448B rows, 64B-aligned tiles)

__device__ __forceinline__ int imin(int a, int b) { return a < b ? a : b; }

__device__ __forceinline__ v8f wmma32(v16h a, v16h b, v8f c) {
    return __builtin_amdgcn_wmma_f32_16x16x32_f16(false, a, false, b, (short)0, c, false, false);
}

// two contiguous 16B chunks -> one 16-half fragment
__device__ __forceinline__ v16h pair16(const _Float16* p0, const _Float16* p1) {
    v8h lo = *(const v8h*)p0;
    v8h hi = *(const v8h*)p1;
    v16h v;
#pragma unroll
    for (int e = 0; e < 8; ++e) { v[e] = lo[e]; v[e + 8] = hi[e]; }
    return v;
}

// ---- Fragment loaders (CDNA5 wave32 16x16x32 f16 layouts, ISA 7.12.2) ----
// A 16x32 (MxK), row-major f16 source: lane half selects K 0-7/16-23 vs 8-15/24-31.
__device__ __forceinline__ v16h load_a_f16(const _Float16* base, int ld, int row0, int rowmax,
                                           int kbase, int lane) {
    int r = imin(row0 + (lane & 15), rowmax);
    int koff = kbase + ((lane & 16) ? 8 : 0);
    const _Float16* rp = base + (size_t)r * ld + koff;
    return pair16(rp, rp + 16);
}

// A 16x32 from f32 (LDS softmax probs): 4x float4 + cvt
__device__ __forceinline__ v16h load_a_f32(const float* base, int ld, int lane) {
    int r = lane & 15;
    int koff = (lane & 16) ? 8 : 0;
    const float* rp = base + (size_t)r * ld + koff;
    float4 f0 = *(const float4*)rp;
    float4 f1 = *(const float4*)(rp + 4);
    float4 f2 = *(const float4*)(rp + 16);
    float4 f3 = *(const float4*)(rp + 20);
    v16h v;
    v[0] = (_Float16)f0.x;  v[1] = (_Float16)f0.y;  v[2] = (_Float16)f0.z;  v[3] = (_Float16)f0.w;
    v[4] = (_Float16)f1.x;  v[5] = (_Float16)f1.y;  v[6] = (_Float16)f1.z;  v[7] = (_Float16)f1.w;
    v[8] = (_Float16)f2.x;  v[9] = (_Float16)f2.y;  v[10] = (_Float16)f2.z; v[11] = (_Float16)f2.w;
    v[12] = (_Float16)f3.x; v[13] = (_Float16)f3.y; v[14] = (_Float16)f3.z; v[15] = (_Float16)f3.w;
    return v;
}

// B 32x16 (KxN) where source is stored [n][k] row-major (transposed weights / vT / keys):
// lane n = lane&15, lane half selects K 0-7/16-23 vs 8-15/24-31 -> two b128 loads.
__device__ __forceinline__ v16h load_bt_f16(const _Float16* base, int ld, int row0, int rowmax,
                                            int kbase, int lane) {
    int r = imin(row0 + (lane & 15), rowmax);
    int kb = kbase + ((lane & 16) ? 8 : 0);
    const _Float16* rp = base + (size_t)r * ld + kb;
    return pair16(rp, rp + 16);
}

// same but f32 source (knn_k), 4x float4 + cvt
__device__ __forceinline__ v16h load_bt_f32(const float* base, int ld, int row0, int kbase,
                                            int lane) {
    const float* rp = base + (size_t)(row0 + (lane & 15)) * ld + kbase + ((lane & 16) ? 8 : 0);
    float4 f0 = *(const float4*)rp;
    float4 f1 = *(const float4*)(rp + 4);
    float4 f2 = *(const float4*)(rp + 16);
    float4 f3 = *(const float4*)(rp + 20);
    v16h v;
    v[0] = (_Float16)f0.x;  v[1] = (_Float16)f0.y;  v[2] = (_Float16)f0.z;  v[3] = (_Float16)f0.w;
    v[4] = (_Float16)f1.x;  v[5] = (_Float16)f1.y;  v[6] = (_Float16)f1.z;  v[7] = (_Float16)f1.w;
    v[8] = (_Float16)f2.x;  v[9] = (_Float16)f2.y;  v[10] = (_Float16)f2.z; v[11] = (_Float16)f2.w;
    v[12] = (_Float16)f3.x; v[13] = (_Float16)f3.y; v[14] = (_Float16)f3.z; v[15] = (_Float16)f3.w;
    return v;
}

// epilogue: uniform fast path for full 16-row tiles (12 of 13)
template <typename F>
__device__ __forceinline__ void store_rows(int rt, int m0, F f) {
    if (rt < 12) {
#pragma unroll
        for (int rr = 0; rr < 8; ++rr) f(rr, rt * 16 + m0 + rr);
    } else {
#pragma unroll
        for (int rr = 0; rr < 8; ++rr) {
            int tk = rt * 16 + m0 + rr;
            if (tk < 196) f(rr, tk);
        }
    }
}

// ---- LayerNorm of a 16-token tile into LDS f16 (block = 256 threads) ----
__device__ __forceinline__ void ln_tile(const float* __restrict__ h, const float* __restrict__ g,
                                        const float* __restrict__ be, int b, int rt,
                                        float (*red)[17], float (*stat)[2],
                                        _Float16 (*yt)[128]) {
    int tid = threadIdx.x;
    int r = tid >> 4, cg = tid & 15;
    int token = rt * 16 + r;
    bool valid = token < 196;
    const float* hrow = h + ((size_t)b * 196 + imin(token, 195)) * 128;
    float xv[8];
    float s = 0.f;
#pragma unroll
    for (int j = 0; j < 8; ++j) { xv[j] = valid ? hrow[cg + 16 * j] : 0.f; s += xv[j]; }
    red[r][cg] = s;
    __syncthreads();
    if (cg == 0) {
        float t = 0.f;
        for (int j = 0; j < 16; ++j) t += red[r][j];
        stat[r][0] = t * (1.f / 128.f);
    }
    __syncthreads();
    float mu = stat[r][0];
    s = 0.f;
#pragma unroll
    for (int j = 0; j < 8; ++j) { float d = xv[j] - mu; s += d * d; }
    red[r][cg] = s;
    __syncthreads();
    if (cg == 0) {
        float t = 0.f;
        for (int j = 0; j < 16; ++j) t += red[r][j];
        stat[r][1] = rsqrtf(t * (1.f / 128.f) + 1e-5f);
    }
    __syncthreads();
    float rstd = stat[r][1];
#pragma unroll
    for (int j = 0; j < 8; ++j) {
        int c = cg + 16 * j;
        yt[r][c] = (_Float16)((xv[j] - mu) * rstd * g[c] + be[c]);
    }
    __syncthreads();
}

// ================= K0: weight prep: Wt[n][k] = (f16) W[k][n] =================
__global__ void transpose_w_kernel(const float* __restrict__ W, _Float16* __restrict__ Wt,
                                   int K, int N) {
    int idx = blockIdx.x * 256 + threadIdx.x;
    if (idx >= K * N) return;
    int n = idx / K, k = idx - n * K;
    Wt[(size_t)n * K + k] = (_Float16)W[(size_t)k * N + n];
}

// ================= K1: patch embed  h = patchify(x) @ patch_w + patch_b =================
__global__ void patch_embed_kernel(const float* __restrict__ x, const _Float16* __restrict__ pwt,
                                   const float* __restrict__ pb, float* __restrict__ h) {
    int b = blockIdx.x, rt = blockIdx.y;
    int lane = threadIdx.x & 31, wv = threadIdx.x >> 5;   // 8 waves, 8 col tiles
    int r = lane & 15;
    int tok = imin(rt * 16 + r, 195);
    int py = tok / 14, px = tok % 14;
    int koff = (lane & 16) ? 8 : 0;
    v8f acc = {};
    for (int kb = 0; kb < 24; ++kb) {
        v16h a;
#pragma unroll
        for (int e = 0; e < 16; ++e) {
            int k = koff + (e < 8 ? e : e + 8);
            int c = kb * 32 + k;                 // 0..767
            int ch = c >> 8, iy = (c >> 4) & 15, ix = c & 15;
            a[e] = (_Float16)x[(((size_t)b * 3 + ch) * 224 + (py * 16 + iy)) * 224 +
                               (px * 16 + ix)];
        }
        v16h bb = load_bt_f16(pwt, 768, wv * 16, 127, kb * 32, lane);
        acc = wmma32(a, bb, acc);
    }
    int ccol = wv * 16 + (lane & 15);
    int m0 = (lane & 16) ? 8 : 0;
    store_rows(rt, m0, [&](int rr, int tk) {
        h[((size_t)b * 196 + tk) * 128 + ccol] = acc[rr] + pb[ccol];
    });
}

// ================= K2: y = LN1(h); qkv = y @ W + b -> q,k (f16), vT (f16 transposed) ==========
__global__ void ln_qkv_kernel(const float* __restrict__ h, const float* __restrict__ g,
                              const float* __restrict__ be, const _Float16* __restrict__ Wt,
                              const float* __restrict__ wb, _Float16* __restrict__ q,
                              _Float16* __restrict__ k, _Float16* __restrict__ vT) {
    __shared__ float red[16][17];
    __shared__ float stat[16][2];
    __shared__ __align__(16) _Float16 yt[16][128];
    int b = blockIdx.x, rt = blockIdx.y;
    ln_tile(h, g, be, b, rt, red, stat, yt);
    int lane = threadIdx.x & 31, wv = threadIdx.x >> 5;
    v16h af[4];
#pragma unroll
    for (int ks = 0; ks < 4; ++ks) af[ks] = load_a_f16(&yt[0][0], 128, 0, 15, ks * 32, lane);
    for (int t3 = 0; t3 < 3; ++t3) {
        int ct = wv + 8 * t3;                     // 24 col tiles of 384
        v8f acc = {};
#pragma unroll
        for (int ks = 0; ks < 4; ++ks) {
            v16h bb = load_bt_f16(Wt, 128, ct * 16, 383, ks * 32, lane);
            acc = wmma32(af[ks], bb, acc);
        }
        int ccol = ct * 16 + (lane & 15);
        int m0 = (lane & 16) ? 8 : 0;
        int which = ccol >> 7, d = ccol & 127;
        float bias = wb[ccol];
        if (which < 2) {
            _Float16* dst = which == 0 ? q : k;
            store_rows(rt, m0, [&](int rr, int tk) {
                dst[((size_t)b * 196 + tk) * 128 + d] = (_Float16)(acc[rr] + bias);
            });
        } else {
            _Float16* vrow = vT + ((size_t)b * 128 + d) * VPAD;
            store_rows(rt, m0, [&](int rr, int tk) { vrow[tk] = (_Float16)(acc[rr] + bias); });
            if (rt == 12)
                for (int tz = 196; tz < VPAD; ++tz) vrow[tz] = (_Float16)0.f;
        }
    }
}

// ================= K3: self attention (softmax(q k^T / sqrt(d)) v) =================
__global__ void attn_kernel(const _Float16* __restrict__ q, const _Float16* __restrict__ k,
                            const _Float16* __restrict__ vT, _Float16* __restrict__ o) {
    __shared__ __align__(16) float sim[16][224];
    int b = blockIdx.x, rt = blockIdx.y;
    int tid = threadIdx.x, lane = tid & 31, wv = tid >> 5;   // 4 waves
    const _Float16* qb = q + (size_t)b * 196 * 128;
    const _Float16* kb = k + (size_t)b * 196 * 128;
    const _Float16* vTb = vT + (size_t)b * 128 * VPAD;
    v16h aq[4];
#pragma unroll
    for (int ks = 0; ks < 4; ++ks) aq[ks] = load_a_f16(qb, 128, rt * 16, 195, ks * 32, lane);
    for (int kt = wv; kt < 13; kt += 4) {
        v8f acc = {};
#pragma unroll
        for (int ks = 0; ks < 4; ++ks) {
            v16h bt = load_bt_f16(kb, 128, kt * 16, 195, ks * 32, lane);
            acc = wmma32(aq[ks], bt, acc);
        }
        int ccol = kt * 16 + (lane & 15);
        int m0 = (lane & 16) ? 8 : 0;
#pragma unroll
        for (int rr = 0; rr < 8; ++rr) sim[m0 + rr][ccol] = acc[rr] * SCALEF;
    }
    __syncthreads();
    if (tid < 16) {
        float m = -1e30f;
        for (int c = 0; c < 196; ++c) m = fmaxf(m, sim[tid][c]);
        float s = 0.f;
        for (int c = 0; c < 196; ++c) { float e = __expf(sim[tid][c] - m); sim[tid][c] = e; s += e; }
        float inv = 1.f / s;
        for (int c = 0; c < 196; ++c) sim[tid][c] *= inv;
        for (int c = 196; c < 224; ++c) sim[tid][c] = 0.f;
    }
    __syncthreads();
    v16h as[7];
#pragma unroll
    for (int ks = 0; ks < 7; ++ks) as[ks] = load_a_f32(&sim[0][ks * 32], 224, lane);
    for (int ct = wv; ct < 8; ct += 4) {
        v8f acc = {};
#pragma unroll
        for (int ks = 0; ks < 7; ++ks) {
            v16h bb = load_bt_f16(vTb, VPAD, ct * 16, 127, ks * 32, lane);
            acc = wmma32(as[ks], bb, acc);
        }
        int ccol = ct * 16 + (lane & 15);
        int m0 = (lane & 16) ? 8 : 0;
        store_rows(rt, m0, [&](int rr, int tk) {
            o[((size_t)b * 196 + tk) * 128 + ccol] = (_Float16)acc[rr];
        });
    }
}

// ================= K4: h += A(f16,[B,196,K]) @ W(K x 128) + bias  (Wt f16 [128][K]) ==========
template <int KST>
__global__ void gemm_res_kernel(const _Float16* __restrict__ A, const _Float16* __restrict__ Wt,
                                const float* __restrict__ bias, float* __restrict__ h) {
    const int K = KST * 32;
    int b = blockIdx.x, rt = blockIdx.y;
    int lane = threadIdx.x & 31, wv = threadIdx.x >> 5;   // 4 waves
    const _Float16* Ab = A + (size_t)b * 196 * K;
    if constexpr (KST <= 4) {
        v16h af[KST];
#pragma unroll
        for (int ks = 0; ks < KST; ++ks) af[ks] = load_a_f16(Ab, K, rt * 16, 195, ks * 32, lane);
        for (int ct = wv; ct < 8; ct += 4) {
            v8f acc = {};
#pragma unroll
            for (int ks = 0; ks < KST; ++ks) {
                v16h bb = load_bt_f16(Wt, K, ct * 16, 127, ks * 32, lane);
                acc = wmma32(af[ks], bb, acc);
            }
            int ccol = ct * 16 + (lane & 15);
            int m0 = (lane & 16) ? 8 : 0;
            float bs = bias[ccol];
            store_rows(rt, m0, [&](int rr, int tk) {
                size_t idx = ((size_t)b * 196 + tk) * 128 + ccol;
                h[idx] = h[idx] + acc[rr] + bs;
            });
        }
    } else {
        for (int ct = wv; ct < 8; ct += 4) {
            v8f acc = {};
            for (int ks = 0; ks < KST; ++ks) {
                v16h a = load_a_f16(Ab, K, rt * 16, 195, ks * 32, lane);
                v16h bb = load_bt_f16(Wt, K, ct * 16, 127, ks * 32, lane);
                acc = wmma32(a, bb, acc);
            }
            int ccol = ct * 16 + (lane & 15);
            int m0 = (lane & 16) ? 8 : 0;
            float bs = bias[ccol];
            store_rows(rt, m0, [&](int rr, int tk) {
                size_t idx = ((size_t)b * 196 + tk) * 128 + ccol;
                h[idx] = h[idx] + acc[rr] + bs;
            });
        }
    }
}

// ================= K5: y = LN2(h); u = gelu(y @ ff1_w + ff1_b) (f16) =================
__global__ void ln_ffn1_kernel(const float* __restrict__ h, const float* __restrict__ g,
                               const float* __restrict__ be, const _Float16* __restrict__ Wt,
                               const float* __restrict__ wb, _Float16* __restrict__ u) {
    __shared__ float red[16][17];
    __shared__ float stat[16][2];
    __shared__ __align__(16) _Float16 yt[16][128];
    int b = blockIdx.x, rt = blockIdx.y;
    ln_tile(h, g, be, b, rt, red, stat, yt);
    int lane = threadIdx.x & 31, wv = threadIdx.x >> 5;
    v16h af[4];
#pragma unroll
    for (int ks = 0; ks < 4; ++ks) af[ks] = load_a_f16(&yt[0][0], 128, 0, 15, ks * 32, lane);
    for (int t4 = 0; t4 < 4; ++t4) {
        int ct = wv + 8 * t4;                     // 32 col tiles of 512
        v8f acc = {};
#pragma unroll
        for (int ks = 0; ks < 4; ++ks) {
            v16h bb = load_bt_f16(Wt, 128, ct * 16, 511, ks * 32, lane);
            acc = wmma32(af[ks], bb, acc);
        }
        int ccol = ct * 16 + (lane & 15);
        int m0 = (lane & 16) ? 8 : 0;
        float bs = wb[ccol];
        store_rows(rt, m0, [&](int rr, int tk) {
            float z = acc[rr] + bs;
            float gz = 0.5f * z * (1.f + tanhf(0.79788456080286535f * (z + 0.044715f * z * z * z)));
            u[((size_t)b * 196 + tk) * 512 + ccol] = (_Float16)gz;
        });
    }
}

// ================= K6: memory (KNN) attention layer =================
__global__ void knn_attn_kernel(const _Float16* __restrict__ q, const _Float16* __restrict__ k,
                                const _Float16* __restrict__ vT, const float* __restrict__ knnk,
                                const float* __restrict__ knnv, _Float16* __restrict__ o) {
    __shared__ __align__(16) float sim[16][224];
    __shared__ float chunk[16][256];
    __shared__ float topv[16][32];
    __shared__ int topi[16][32];
    __shared__ float omem[16][128];
    int b = blockIdx.x, rt = blockIdx.y;
    int tid = threadIdx.x, lane = tid & 31, wv = tid >> 5;   // 4 waves
    const _Float16* qb = q + (size_t)b * 196 * 128;
    const _Float16* kb = k + (size_t)b * 196 * 128;
    const _Float16* vTb = vT + (size_t)b * 128 * VPAD;
    const float* Kb = knnk + (size_t)b * 8192 * 128;
    const float* Vb = knnv + (size_t)b * 8192 * 128;

    // q tile fragments: reused for all 13 self tiles + 512 memory tiles
    v16h aq[4];
#pragma unroll
    for (int ks = 0; ks < 4; ++ks) aq[ks] = load_a_f16(qb, 128, rt * 16, 195, ks * 32, lane);

    // self similarities
    for (int kt = wv; kt < 13; kt += 4) {
        v8f acc = {};
#pragma unroll
        for (int ks = 0; ks < 4; ++ks) {
            v16h bt = load_bt_f16(kb, 128, kt * 16, 195, ks * 32, lane);
            acc = wmma32(aq[ks], bt, acc);
        }
        int ccol = kt * 16 + (lane & 15);
        int m0 = (lane & 16) ? 8 : 0;
#pragma unroll
        for (int rr = 0; rr < 8; ++rr) sim[m0 + rr][ccol] = acc[rr] * SCALEF;
    }
    for (int idx = tid; idx < 16 * 32; idx += 128) {
        (&topv[0][0])[idx] = -1e30f;
        (&topi[0][0])[idx] = 0;
    }
    __syncthreads();

    // stream 8192 memory keys in 32 chunks of 256, maintain per-row top-32
    float curmin = -1e30f;
    int minpos = 0;
    for (int ch = 0; ch < 32; ++ch) {
        if (ch + 1 < 32) {   // prefetch next chunk of knn_k (global_prefetch_b8)
            const float* pf = Kb + ((size_t)(ch + 1) * 256 + tid * 2) * 128;
            __builtin_prefetch((const void*)pf, 0, 1);
        }
        for (int j = 0; j < 4; ++j) {
            int kt2 = wv * 4 + j;                 // 16 tiles per chunk
            int kg = ch * 256 + kt2 * 16;
            v8f acc = {};
#pragma unroll
            for (int ks = 0; ks < 4; ++ks) {
                v16h bt = load_bt_f32(Kb, 128, kg, ks * 32, lane);
                acc = wmma32(aq[ks], bt, acc);
            }
            int ccol = kt2 * 16 + (lane & 15);
            int m0 = (lane & 16) ? 8 : 0;
#pragma unroll
            for (int rr = 0; rr < 8; ++rr) chunk[m0 + rr][ccol] = acc[rr] * SCALEF;
        }
        __syncthreads();
        if (tid < 16) {
            for (int j = 0; j < 256; ++j) {
                float val = chunk[tid][j];
                if (val > curmin) {
                    topv[tid][minpos] = val;
                    topi[tid][minpos] = ch * 256 + j;
                    curmin = topv[tid][0]; minpos = 0;
                    for (int t2 = 1; t2 < 32; ++t2) {
                        float tv = topv[tid][t2];
                        if (tv < curmin) { curmin = tv; minpos = t2; }
                    }
                }
            }
        }
        __syncthreads();
    }

    // joint softmax over [top32 | 196 self]
    if (tid < 16) {
        float m = -1e30f;
        for (int j = 0; j < 32; ++j) m = fmaxf(m, topv[tid][j]);
        for (int c = 0; c < 196; ++c) m = fmaxf(m, sim[tid][c]);
        float s = 0.f;
        for (int j = 0; j < 32; ++j) { float e = __expf(topv[tid][j] - m); topv[tid][j] = e; s += e; }
        for (int c = 0; c < 196; ++c) { float e = __expf(sim[tid][c] - m); sim[tid][c] = e; s += e; }
        float inv = 1.f / s;
        for (int j = 0; j < 32; ++j) topv[tid][j] *= inv;
        for (int c = 0; c < 196; ++c) sim[tid][c] *= inv;
        for (int c = 196; c < 224; ++c) sim[tid][c] = 0.f;
    }
    __syncthreads();

    // memory-value gather: omem[r][d] = sum_j am[r][j] * knn_v[idx[r][j]][d]
    {
        int d = tid;   // 128 threads = 128 dims
        for (int r = 0; r < 16; ++r) {
            float a = 0.f;
#pragma unroll 4
            for (int j = 0; j < 32; ++j) a += topv[r][j] * Vb[(size_t)topi[r][j] * 128 + d];
            omem[r][d] = a;
        }
    }
    __syncthreads();

    // o = al @ v + omem
    v16h as[7];
#pragma unroll
    for (int ks = 0; ks < 7; ++ks) as[ks] = load_a_f32(&sim[0][ks * 32], 224, lane);
    for (int ct = wv; ct < 8; ct += 4) {
        v8f acc = {};
#pragma unroll
        for (int ks = 0; ks < 7; ++ks) {
            v16h bb = load_bt_f16(vTb, VPAD, ct * 16, 127, ks * 32, lane);
            acc = wmma32(as[ks], bb, acc);
        }
        int ccol = ct * 16 + (lane & 15);
        int m0 = (lane & 16) ? 8 : 0;
        store_rows(rt, m0, [&](int rr, int tk) {
            o[((size_t)b * 196 + tk) * 128 + ccol] = (_Float16)(acc[rr] + omem[m0 + rr][ccol]);
        });
    }
}

// ================= K7: feats = mean_n LN(h); out = feats @ head_w + head_b =================
__global__ void head_kernel(const float* __restrict__ h, const float* __restrict__ g,
                            const float* __restrict__ be, const float* __restrict__ hw,
                            const float* __restrict__ hb, float* __restrict__ out) {
    __shared__ float red[128];
    __shared__ float feats[128];
    int b = blockIdx.x, tid = threadIdx.x;   // 128 threads
    float acc = 0.f;
    for (int n = 0; n < 196; ++n) {
        float xv = h[((size_t)b * 196 + n) * 128 + tid];
        red[tid] = xv;
        __syncthreads();
        for (int s = 64; s > 0; s >>= 1) { if (tid < s) red[tid] += red[tid + s]; __syncthreads(); }
        float mu = red[0] * (1.f / 128.f);
        __syncthreads();
        float dv = xv - mu;
        red[tid] = dv * dv;
        __syncthreads();
        for (int s = 64; s > 0; s >>= 1) { if (tid < s) red[tid] += red[tid + s]; __syncthreads(); }
        float var = red[0] * (1.f / 128.f);
        __syncthreads();
        acc += dv * rsqrtf(var + 1e-5f) * g[tid] + be[tid];
    }
    feats[tid] = acc * (1.f / 196.f);
    __syncthreads();
    for (int oo = tid; oo < 1000; oo += 128) {
        float a = hb[oo];
        for (int kk = 0; kk < 128; ++kk) a += feats[kk] * hw[kk * 1000 + oo];
        out[(size_t)b * 1000 + oo] = a;
    }
}

extern "C" void kernel_launch(void* const* d_in, const int* in_sizes, int n_in, void* d_out,
                              int out_size, void* d_ws, size_t ws_size, hipStream_t stream) {
    (void)in_sizes; (void)n_in; (void)out_size; (void)ws_size;
    const float* x       = (const float*)d_in[0];
    const float* patch_w = (const float*)d_in[1];
    const float* patch_b = (const float*)d_in[2];
    const float* ln1_g   = (const float*)d_in[3];
    const float* ln1_b   = (const float*)d_in[4];
    const float* qkv_w   = (const float*)d_in[5];
    const float* qkv_b   = (const float*)d_in[6];
    const float* out_w   = (const float*)d_in[7];
    const float* out_b   = (const float*)d_in[8];
    const float* ln2_g   = (const float*)d_in[9];
    const float* ln2_b   = (const float*)d_in[10];
    const float* ff1_w   = (const float*)d_in[11];
    const float* ff1_b   = (const float*)d_in[12];
    const float* ff2_w   = (const float*)d_in[13];
    const float* ff2_b   = (const float*)d_in[14];
    const float* lnf_g   = (const float*)d_in[15];
    const float* lnf_b   = (const float*)d_in[16];
    const float* knn_k   = (const float*)d_in[17];
    const float* knn_v   = (const float*)d_in[18];
    const float* head_w  = (const float*)d_in[19];
    const float* head_b  = (const float*)d_in[20];

    const size_t HB = (size_t)32 * 196 * 128;   // activation elements
    char* ws = (char*)d_ws;
    float*    h   = (float*)ws;                              // HB f32
    _Float16* q   = (_Float16*)(ws + HB * 4);                // HB f16
    _Float16* kf  = q + HB;
    _Float16* vT  = kf + HB;                                 // 32*128*VPAD f16
    _Float16* o   = vT + (size_t)32 * 128 * VPAD;
    _Float16* u   = o + HB;                                  // 4*HB f16
    _Float16* wpt = u + 4 * HB;                              // patch_w^T  128x768
    _Float16* wqk = wpt + (size_t)128 * 768;                 // qkv_w^T    7x 384x128
    _Float16* wot = wqk + (size_t)7 * 384 * 128;             // out_w^T    7x 128x128
    _Float16* wf1 = wot + (size_t)7 * 128 * 128;             // ff1_w^T    7x 512x128
    _Float16* wf2 = wf1 + (size_t)7 * 512 * 128;             // ff2_w^T    7x 128x512
    // total ws used ~= 19.2 MB

    // one-time-per-launch weight transpose+f16 (tiny vs 128MB knn stream)
    transpose_w_kernel<<<(768 * 128 + 255) / 256, 256, 0, stream>>>(patch_w, wpt, 768, 128);
    for (int i = 0; i < 7; ++i) {
        transpose_w_kernel<<<(128 * 384 + 255) / 256, 256, 0, stream>>>(
            qkv_w + (size_t)i * 128 * 384, wqk + (size_t)i * 384 * 128, 128, 384);
        transpose_w_kernel<<<(128 * 128 + 255) / 256, 256, 0, stream>>>(
            out_w + (size_t)i * 128 * 128, wot + (size_t)i * 128 * 128, 128, 128);
        transpose_w_kernel<<<(128 * 512 + 255) / 256, 256, 0, stream>>>(
            ff1_w + (size_t)i * 128 * 512, wf1 + (size_t)i * 512 * 128, 128, 512);
        transpose_w_kernel<<<(512 * 128 + 255) / 256, 256, 0, stream>>>(
            ff2_w + (size_t)i * 512 * 128, wf2 + (size_t)i * 128 * 512, 512, 128);
    }

    dim3 g2(32, 13);
    patch_embed_kernel<<<g2, 256, 0, stream>>>(x, wpt, patch_b, h);
    for (int i = 0; i < 7; ++i) {
        ln_qkv_kernel<<<g2, 256, 0, stream>>>(h, ln1_g + i * 128, ln1_b + i * 128,
                                              wqk + (size_t)i * 384 * 128, qkv_b + i * 384,
                                              q, kf, vT);
        if (i == 6)
            knn_attn_kernel<<<g2, 128, 0, stream>>>(q, kf, vT, knn_k, knn_v, o);
        else
            attn_kernel<<<g2, 128, 0, stream>>>(q, kf, vT, o);
        gemm_res_kernel<4><<<g2, 128, 0, stream>>>(o, wot + (size_t)i * 128 * 128,
                                                   out_b + i * 128, h);
        ln_ffn1_kernel<<<g2, 256, 0, stream>>>(h, ln2_g + i * 128, ln2_b + i * 128,
                                               wf1 + (size_t)i * 512 * 128, ff1_b + i * 512, u);
        gemm_res_kernel<16><<<g2, 128, 0, stream>>>(u, wf2 + (size_t)i * 128 * 512,
                                                    ff2_b + i * 128, h);
    }
    head_kernel<<<32, 128, 0, stream>>>(h, lnf_g, lnf_b, head_w, head_b, (float*)d_out);
}
